// VocabParallelOutput_18030272709050
// MI455X (gfx1250) — matrix operand order repaired and checked by reading, not compiled, not run on hardware
//
#include <hip/hip_runtime.h>
#include <hip/hip_bf16.h>
#include <math.h>

// ---------------------------------------------------------------------------
// Fused vocab-parallel cross-entropy for MI455X (gfx1250).
//   logits = x[4096,1024] @ W[32000,1024]^T   (bf16 WMMA, f32 accum)
//   loss[row] = log(sum_exp(logits - max)) - (logits[target] - max)
// Kernel 0 (if ws fits): repack W f32 -> bf16 in WMMA B-fragment order.
// Kernel 1: flash-style GEMM + online logsumexp over V chunks -> partials.
//           (packed path: 2 N-tiles per A fragment -> 1 ds_b128 per WMMA)
// Kernel 2: combine partials across V blocks.
// ---------------------------------------------------------------------------

typedef __bf16 bf16_t;
typedef __attribute__((ext_vector_type(16))) bf16_t v16bf;
typedef __attribute__((ext_vector_type(8)))  bf16_t v8bf;
typedef __attribute__((ext_vector_type(4)))  bf16_t v4bf;
typedef __attribute__((ext_vector_type(8)))  float  v8f;

#define HDIM     1024
#define VDIM     32000
#define MROWS    4096                    // S*B
#define M_TILE   128
#define M_BLOCKS (MROWS / M_TILE)        // 32
#define V_CHUNK  1280
#define V_BLOCKS (VDIM / V_CHUNK)        // 25
#define NCHUNK   128                     // columns per inner chunk (8 N-tiles x 16)
#define NC_ITERS (V_CHUNK / NCHUNK)      // 10
#define KSTEPS   (HDIM / 32)             // 32
#define LDS_ROW  1032                    // bf16/row, padded: 516 dwords -> conflict-free b128
#define LDS_BYTES (M_TILE * LDS_ROW * (int)sizeof(bf16_t))   // 264192 B < 320KB/WGP

// Packed-W geometry: fragment = 16 bf16 per lane; 32 lanes per (tile,kstep);
// 32 ksteps per 16-column tile.  Total = VDIM*HDIM bf16 = 65.536 MB.
#define FRAG_ELEMS   16
#define KBLK_ELEMS   (32 * FRAG_ELEMS)            // 512
#define TILE_ELEMS   (KSTEPS * KBLK_ELEMS)        // 16384

// ---------------------------------------------------------------------------
// Kernel 0: W[v][h] f32 -> packed bf16 fragments (reads fully coalesced).
// ---------------------------------------------------------------------------
__global__ __launch_bounds__(256)
void pack_w_kernel(const float* __restrict__ w, bf16_t* __restrict__ wpk)
{
    const int t   = blockIdx.x * 256 + threadIdx.x;   // 0 .. VDIM*HDIM/8 - 1
    const int row = t >> 7;                            // / (HDIM/8)
    const int k   = (t & 127) << 3;                    // h offset, step 8

    const float4 a = *(const float4*)(w + (size_t)row * HDIM + k);
    const float4 b = *(const float4*)(w + (size_t)row * HDIM + k + 4);
    v8bf v;
    v[0] = (bf16_t)a.x; v[1] = (bf16_t)a.y; v[2] = (bf16_t)a.z; v[3] = (bf16_t)a.w;
    v[4] = (bf16_t)b.x; v[5] = (bf16_t)b.y; v[6] = (bf16_t)b.z; v[7] = (bf16_t)b.w;

    // Fragment coordinates: lane (L&15)=column-in-tile, (L>>4)=K-half.
    const int kstep = k >> 5;
    const int khalf = (k >> 4) & 1;
    const int lane  = (khalf << 4) | (row & 15);
    const int tile  = row >> 4;
    const size_t dst = ((size_t)(tile * KSTEPS + kstep) * 32 + lane) * FRAG_ELEMS
                       + (size_t)(k & 15);            // k%16 = position in fragment
    *(v8bf*)(wpk + dst) = v;
}

// ---------------------------------------------------------------------------
// Shared device helpers for kernel 1.
// ---------------------------------------------------------------------------
__device__ __forceinline__ void stage_x_tile(const float* __restrict__ x,
                                             bf16_t* __restrict__ A,
                                             int m0, int tid)
{
    for (int i = tid; i < M_TILE * (HDIM / 4); i += 256) {
        const int row = i >> 8;
        const int c4  = (i & 255) << 2;
        const float4 xv = *(const float4*)(x + (size_t)(m0 + row) * HDIM + c4);
        v4bf bv;
        bv.x = (bf16_t)xv.x; bv.y = (bf16_t)xv.y;
        bv.z = (bf16_t)xv.z; bv.w = (bf16_t)xv.w;
        *(v4bf*)(A + row * LDS_ROW + c4) = bv;
    }
}

__device__ __forceinline__ void lse_update(const v8f& acc, int mycol,
                                           const int* t8, float* m8,
                                           float* s8, float* p8)
{
    #pragma unroll
    for (int r = 0; r < 8; ++r) {
        const float l  = acc[r];
        const float nm = fmaxf(m8[r], l);
        s8[r] = s8[r] * __expf(m8[r] - nm) + __expf(l - nm);
        m8[r] = nm;
        if (mycol == t8[r]) p8[r] = l;   // raw logit at target column
    }
}

__device__ __forceinline__ void reduce_and_store(char* smem_raw, int tid,
                                                 int wave, int kh, int cL,
                                                 int m0, int vb,
                                                 const float* m8, const float* s8,
                                                 const float* p8,
                                                 float* __restrict__ pm,
                                                 float* __restrict__ ps,
                                                 float* __restrict__ pt)
{
    __syncthreads();                               // A image no longer needed
    float* redm = (float*)smem_raw;                // [128][16]
    float* reds = redm + M_TILE * 16;
    float* redt = reds + M_TILE * 16;
    #pragma unroll
    for (int r = 0; r < 8; ++r) {
        const int rl = wave * 16 + kh * 8 + r;
        redm[rl * 16 + cL] = m8[r];
        reds[rl * 16 + cL] = s8[r];
        redt[rl * 16 + cL] = p8[r];
    }
    __syncthreads();

    if (tid < M_TILE) {
        float M = -INFINITY, T = -INFINITY, S = 0.0f;
        #pragma unroll
        for (int i = 0; i < 16; ++i) M = fmaxf(M, redm[tid * 16 + i]);
        #pragma unroll
        for (int i = 0; i < 16; ++i) S += reds[tid * 16 + i] * __expf(redm[tid * 16 + i] - M);
        #pragma unroll
        for (int i = 0; i < 16; ++i) T = fmaxf(T, redt[tid * 16 + i]);
        const int rg = m0 + tid;
        pm[rg * V_BLOCKS + vb] = M;
        ps[rg * V_BLOCKS + vb] = S;
        pt[rg * V_BLOCKS + vb] = T;   // raw target logit iff target in this V block
    }
}

// ---------------------------------------------------------------------------
// Kernel 1a (packed W): dual N-tile per wave -> each A fragment feeds 2 WMMAs.
// Steady state per WMMA: 1x ds_load_b128 (A) + 2x global_load_b128 (B, 1KB
// coalesced per wave-pair) + 1x v_wmma_f32_16x16x32_bf16.
// ---------------------------------------------------------------------------
__global__ __launch_bounds__(256, 1)
void ce_partial_pre_kernel(const float*  __restrict__ x,
                           const bf16_t* __restrict__ wpk,
                           const int*    __restrict__ tgt,
                           float* __restrict__ pm,
                           float* __restrict__ ps,
                           float* __restrict__ pt)
{
    extern __shared__ char smem_raw[];
    bf16_t* A = (bf16_t*)smem_raw;

    const int vb   = blockIdx.x;
    const int mb   = blockIdx.y;
    const int m0   = mb * M_TILE;
    const int v0   = vb * V_CHUNK;
    const int tid  = threadIdx.x;
    const int lane = tid & 31;
    const int wave = tid >> 5;           // owns rows [wave*16, wave*16+16)
    const int cL   = lane & 15;
    const int kh   = lane >> 4;

    stage_x_tile(x, A, m0, tid);
    __syncthreads();

    int   t8[8];
    float m8[8], s8[8], p8[8];
    #pragma unroll
    for (int r = 0; r < 8; ++r) {
        t8[r] = tgt[m0 + wave * 16 + kh * 8 + r];
        m8[r] = -INFINITY; s8[r] = 0.0f; p8[r] = -INFINITY;
    }

    const bf16_t* Arow = A + (wave * 16 + cL) * LDS_ROW;

    for (int nc = 0; nc < NC_ITERS; ++nc) {
        const int nbase = v0 + nc * NCHUNK;
        #pragma unroll 1
        for (int np = 0; np < 4; ++np) {           // 4 pairs = 8 N-tiles
            const int col0 = nbase + np * 32;
            const bf16_t* f0 = wpk + (size_t)(col0 >> 4) * TILE_ELEMS
                                   + (size_t)lane * FRAG_ELEMS;
            const bf16_t* f1 = f0 + TILE_ELEMS;

            v8f acc0 = {0.f, 0.f, 0.f, 0.f, 0.f, 0.f, 0.f, 0.f};
            v8f acc1 = {0.f, 0.f, 0.f, 0.f, 0.f, 0.f, 0.f, 0.f};

            #pragma unroll 2
            for (int k = 0; k < HDIM; k += 32) {
                // One A fragment serves both N-tiles.
                const v8bf alo = *(const v8bf*)(Arow + k + kh * 16);
                const v8bf ahi = *(const v8bf*)(Arow + k + kh * 16 + 8);
                const v16bf a = __builtin_shufflevector(alo, ahi,
                        0, 1, 2, 3, 4, 5, 6, 7, 8, 9, 10, 11, 12, 13, 14, 15);

                const bf16_t* p0 = f0 + (size_t)(k >> 5) * KBLK_ELEMS;
                const bf16_t* p1 = f1 + (size_t)(k >> 5) * KBLK_ELEMS;
                const v16bf b0 = __builtin_shufflevector(
                        *(const v8bf*)p0, *(const v8bf*)(p0 + 8),
                        0, 1, 2, 3, 4, 5, 6, 7, 8, 9, 10, 11, 12, 13, 14, 15);
                const v16bf b1 = __builtin_shufflevector(
                        *(const v8bf*)p1, *(const v8bf*)(p1 + 8),
                        0, 1, 2, 3, 4, 5, 6, 7, 8, 9, 10, 11, 12, 13, 14, 15);
                if (k + 32 < HDIM) {
                    __builtin_prefetch(p0 + KBLK_ELEMS, 0, 3);
                    __builtin_prefetch(p1 + KBLK_ELEMS, 0, 3);
                }

                acc0 = __builtin_amdgcn_wmma_f32_16x16x32_bf16(
                        false, a, false, b0, (short)0, acc0, false, false);
                acc1 = __builtin_amdgcn_wmma_f32_16x16x32_bf16(
                        false, a, false, b1, (short)0, acc1, false, false);
            }

            lse_update(acc0, col0 + cL,      t8, m8, s8, p8);
            lse_update(acc1, col0 + 16 + cL, t8, m8, s8, p8);
        }
    }

    reduce_and_store(smem_raw, tid, wave, kh, cL, m0, vb, m8, s8, p8, pm, ps, pt);
}

// ---------------------------------------------------------------------------
// Kernel 1b (fallback, ws too small): convert W f32 -> bf16 on the fly.
// ---------------------------------------------------------------------------
__global__ __launch_bounds__(256, 1)
void ce_partial_fb_kernel(const float* __restrict__ x,
                          const float* __restrict__ w,
                          const int*   __restrict__ tgt,
                          float* __restrict__ pm,
                          float* __restrict__ ps,
                          float* __restrict__ pt)
{
    extern __shared__ char smem_raw[];
    bf16_t* A = (bf16_t*)smem_raw;

    const int vb   = blockIdx.x;
    const int mb   = blockIdx.y;
    const int m0   = mb * M_TILE;
    const int v0   = vb * V_CHUNK;
    const int tid  = threadIdx.x;
    const int lane = tid & 31;
    const int wave = tid >> 5;
    const int cL   = lane & 15;
    const int kh   = lane >> 4;

    stage_x_tile(x, A, m0, tid);
    __syncthreads();

    int   t8[8];
    float m8[8], s8[8], p8[8];
    #pragma unroll
    for (int r = 0; r < 8; ++r) {
        t8[r] = tgt[m0 + wave * 16 + kh * 8 + r];
        m8[r] = -INFINITY; s8[r] = 0.0f; p8[r] = -INFINITY;
    }

    const bf16_t* Arow = A + (wave * 16 + cL) * LDS_ROW;

    for (int nc = 0; nc < NC_ITERS; ++nc) {
        const int nbase = v0 + nc * NCHUNK;
        for (int nt = 0; nt < 8; ++nt) {
            const int col0 = nbase + nt * 16;
            const float* wp = w + (size_t)(col0 + cL) * HDIM + kh * 16;

            v8f acc = {0.f, 0.f, 0.f, 0.f, 0.f, 0.f, 0.f, 0.f};
            #pragma unroll 4
            for (int k = 0; k < HDIM; k += 32) {
                const v8bf alo = *(const v8bf*)(Arow + k + kh * 16);
                const v8bf ahi = *(const v8bf*)(Arow + k + kh * 16 + 8);
                const v16bf a = __builtin_shufflevector(alo, ahi,
                        0, 1, 2, 3, 4, 5, 6, 7, 8, 9, 10, 11, 12, 13, 14, 15);

                const float4 b0 = *(const float4*)(wp + k);
                const float4 b1 = *(const float4*)(wp + k + 4);
                const float4 b2 = *(const float4*)(wp + k + 8);
                const float4 b3 = *(const float4*)(wp + k + 12);
                if (k + 32 < HDIM) __builtin_prefetch(wp + k + 32, 0, 3);

                v16bf b;
                b[0]  = (bf16_t)b0.x; b[1]  = (bf16_t)b0.y;
                b[2]  = (bf16_t)b0.z; b[3]  = (bf16_t)b0.w;
                b[4]  = (bf16_t)b1.x; b[5]  = (bf16_t)b1.y;
                b[6]  = (bf16_t)b1.z; b[7]  = (bf16_t)b1.w;
                b[8]  = (bf16_t)b2.x; b[9]  = (bf16_t)b2.y;
                b[10] = (bf16_t)b2.z; b[11] = (bf16_t)b2.w;
                b[12] = (bf16_t)b3.x; b[13] = (bf16_t)b3.y;
                b[14] = (bf16_t)b3.z; b[15] = (bf16_t)b3.w;

                acc = __builtin_amdgcn_wmma_f32_16x16x32_bf16(
                        false, a, false, b, (short)0, acc, false, false);
            }

            lse_update(acc, col0 + cL, t8, m8, s8, p8);
        }
    }

    reduce_and_store(smem_raw, tid, wave, kh, cL, m0, vb, m8, s8, p8, pm, ps, pt);
}

// ---------------------------------------------------------------------------
// Kernel 2: combine per-V-block partials.
// ---------------------------------------------------------------------------
__global__ __launch_bounds__(256)
void ce_combine_kernel(const int*   __restrict__ tgt,
                       const float* __restrict__ pm,
                       const float* __restrict__ ps,
                       const float* __restrict__ pt,
                       float* __restrict__ out)
{
    const int row = blockIdx.x * 256 + threadIdx.x;
    if (row >= MROWS) return;

    float M = -INFINITY;
    #pragma unroll
    for (int b = 0; b < V_BLOCKS; ++b) M = fmaxf(M, pm[row * V_BLOCKS + b]);
    float S = 0.0f;
    #pragma unroll
    for (int b = 0; b < V_BLOCKS; ++b)
        S += ps[row * V_BLOCKS + b] * __expf(pm[row * V_BLOCKS + b] - M);

    const int   t    = tgt[row];
    const float traw = pt[row * V_BLOCKS + (t / V_CHUNK)];
    // loss = log(sum exp(l - M)) - (l_t - M)
    out[row] = logf(S) + M - traw;
}

extern "C" void kernel_launch(void* const* d_in, const int* in_sizes, int n_in,
                              void* d_out, int out_size, void* d_ws, size_t ws_size,
                              hipStream_t stream) {
    const float* x   = (const float*)d_in[0];   // [2048,2,1024] f32
    const float* w   = (const float*)d_in[1];   // [32000,1024] f32
    const int*   tgt = (const int*)  d_in[2];   // [2048,2]
    float*       out = (float*)d_out;           // [2048,2] f32

    const size_t packed_bytes = (size_t)VDIM * HDIM * sizeof(bf16_t);  // 65.536 MB
    const size_t part_bytes   = (size_t)3 * MROWS * V_BLOCKS * sizeof(float);
    const bool   pre          = ws_size >= packed_bytes + part_bytes;  // fixed per run

    dim3 grid(V_BLOCKS, M_BLOCKS);   // 25 x 32 = 800 workgroups

    if (pre) {
        bf16_t* wpk = (bf16_t*)d_ws;
        float*  pm  = (float*)((char*)d_ws + packed_bytes);
        float*  ps  = pm + (size_t)MROWS * V_BLOCKS;
        float*  pt  = ps + (size_t)MROWS * V_BLOCKS;
        // Repack W: read 131MB + write 65.5MB ~ 8.5us at 23.3 TB/s, amortized
        // over the 32 row-blocks that each re-read W through L2.
        pack_w_kernel<<<(VDIM * (HDIM / 8)) / 256, 256, 0, stream>>>(w, wpk);
        ce_partial_pre_kernel<<<grid, 256, LDS_BYTES, stream>>>(
                x, wpk, tgt, pm, ps, pt);
        ce_combine_kernel<<<MROWS / 256, 256, 0, stream>>>(tgt, pm, ps, pt, out);
    } else {
        float* pm = (float*)d_ws;
        float* ps = pm + (size_t)MROWS * V_BLOCKS;
        float* pt = ps + (size_t)MROWS * V_BLOCKS;
        ce_partial_fb_kernel<<<grid, 256, LDS_BYTES, stream>>>(
                x, w, tgt, pm, ps, pt);
        ce_combine_kernel<<<MROWS / 256, 256, 0, stream>>>(tgt, pm, ps, pt, out);
    }

    (void)in_sizes; (void)n_in; (void)out_size;
}
